// SCNN_Model_LIF_hh_41291815584042
// MI455X (gfx1250) — compile-verified, matrix operand order checked by reading, not run on hardware
//
#include <hip/hip_runtime.h>

typedef __attribute__((ext_vector_type(16))) _Float16 v16h;
typedef __attribute__((ext_vector_type(8)))  float    v8f;

#define THRESH 0.3f
#define DECAY  0.2f
#define TW     10

// K-within-32-chunk mapping for 16-bit WMMA A/B fragments (ISA 7.12.2):
// element e (0..15), half-wave hi (lane>=16): e<8 -> K = e + 8*hi ; e>=8 -> K = (e+8) + 8*hi
__device__ __forceinline__ int klocal_map(int e, int hi) {
    return (e < 8 ? e : e + 8) + 8 * hi;
}

// im2col LDS offset for reduction index k (k = cin*9 + tap), relative to
// base = y*20 + x in the zero-padded [10][20][20] f16 tile.
#define OFFK(k) (((k) / 9) * 400 + ((((k) % 9) / 3) * 20) + (((k) % 9) % 3))

// ---------------------------------------------------------------------------
// Kernel 0: pack w2 (3,10,10,3,3) into WMMA-B f16 fragments.
// Layout: [nt(2)][kk(3)][lane(32)][e(16)] ; logical B[k][n], n = gate*10 + oc,
// k = cin*9 + tap, padded K=96 (zeros for k>=90 -> lets A skip K-guards), N=32.
// ---------------------------------------------------------------------------
__global__ void pack_w2_kernel(const float* __restrict__ w2,
                               _Float16* __restrict__ wpack) {
    int idx = blockIdx.x * blockDim.x + threadIdx.x;
    if (idx >= 3072) return;
    int e    = idx & 15;
    int lane = (idx >> 4) & 31;
    int kk   = (idx >> 9) % 3;
    int nt   = idx / 1536;
    int hi   = lane >> 4;
    int ncol = nt * 16 + (lane & 15);
    int k    = kk * 32 + klocal_map(e, hi);
    float v = 0.f;
    if (ncol < 30 && k < 90) {
        int g = ncol / 10, oc = ncol % 10;
        int cin = k / 9, tap = k % 9;
        v = w2[((g * 10 + oc) * 10 + cin) * 9 + tap];
    }
    wpack[idx] = (_Float16)v;
}

// ---------------------------------------------------------------------------
// Kernel 1: layer-1 conv drive + LIF over all 10 steps, fully in registers.
// One thread = one (image, channel, 2x2 pooled block). Emits pooled spike-sum
// s[b][t][c][y][x] (f16, exact: multiples of 0.25 in [0,4]).
// ---------------------------------------------------------------------------
__global__ void __launch_bounds__(256)
layer1_lif_kernel(const float* __restrict__ x,
                  const float* __restrict__ w1,
                  const float* __restrict__ b1,
                  const float* __restrict__ lif_w,
                  const float* __restrict__ lif_b,
                  _Float16* __restrict__ s_out) {
    int idx = blockIdx.x * blockDim.x + threadIdx.x;
    if (idx >= 256 * 10 * 324) return;
    int b  = idx / (10 * 324);
    int r  = idx % (10 * 324);
    int c  = r / 324;
    int q  = r % 324;
    int by = q / 18, bx = q % 18;

    // 4x4 input patch (zero-padded) covering the 2x2 output block's receptive field
    float patch[4][4];
    const float* xb = x + b * (36 * 36);
    #pragma unroll
    for (int dy = 0; dy < 4; ++dy)
        #pragma unroll
        for (int dx = 0; dx < 4; ++dx) {
            int yy = 2 * by - 1 + dy, xx = 2 * bx - 1 + dx;
            patch[dy][dx] = (yy >= 0 && yy < 36 && xx >= 0 && xx < 36)
                                ? xb[yy * 36 + xx] : 0.f;
        }

    // time-invariant conv drives: 4 pixels x 3 gates
    float drive[4][3];
    #pragma unroll
    for (int g = 0; g < 3; ++g) {
        float wk[9];
        #pragma unroll
        for (int t9 = 0; t9 < 9; ++t9)
            wk[t9] = w1[(g * 10 + c) * 9 + t9];   // (3,10,1,3,3)
        float bias = b1[g * 10 + c];
        #pragma unroll
        for (int p = 0; p < 4; ++p) {
            int py = p >> 1, px = p & 1;
            float acc = bias;
            #pragma unroll
            for (int ky = 0; ky < 3; ++ky)
                #pragma unroll
                for (int kx = 0; kx < 3; ++kx)
                    acc += wk[ky * 3 + kx] * patch[py + ky][px + kx];
            drive[p][g] = acc;
        }
    }

    float lw0 = lif_w[0], lw1 = lif_w[1], lw2 = lif_w[2], lb = lif_b[0];
    float mem[4][4], spk[4][4];
    #pragma unroll
    for (int p = 0; p < 4; ++p)
        #pragma unroll
        for (int h = 0; h < 4; ++h) { mem[p][h] = 0.f; spk[p][h] = 0.f; }

    for (int t = 0; t < TW; ++t) {
        float cnt = 0.f;
        #pragma unroll
        for (int p = 0; p < 4; ++p) {
            float inner = lw0 * mem[p][0] + lw1 * mem[p][1] + lw2 * mem[p][2] + lb;
            #pragma unroll
            for (int h = 0; h < 4; ++h) {
                float d  = (h < 3) ? drive[p][h] : inner;
                float mn = mem[p][h] * DECAY * (1.f - spk[p][h]) + d;
                float sv = (mn > THRESH) ? 1.f : 0.f;
                mem[p][h] = mn; spk[p][h] = sv;
                cnt += sv;
            }
        }
        // sum over hh of 2x2-average == total spike count / 4 (exact in f16)
        s_out[(((b * TW + t) * 10 + c) * 18 + by) * 18 + bx] = (_Float16)(cnt * 0.25f);
    }
}

// ---------------------------------------------------------------------------
// Kernel 2: layer-2, one workgroup per image, all temporal state LDS-resident.
// Per step: stage s_t halo tile -> implicit-GEMM conv via v_wmma f16
// (M=324 w/ overlapped last tile, N=30->32, K=90->96) -> LIF + pool + h1.
// One job per M-tile computes BOTH N-tiles: the 48 A gathers are done once
// and reused by 6 WMMAs (2 accumulators), halving the dominant DS cost.
// No per-element guards: B is zero-padded in K, last M-tile overlaps (benign
// identical-value race), im2col offsets hoisted to 48 loop-invariant VGPRs.
// LDS: 8192 s_tile + 43008 drives + 51840 mem + 13056 spk + 12960 h1
//    + 6144 Bfrag + 128 bias = 135328 B  (< 320KB WGP LDS)
// ---------------------------------------------------------------------------
__global__ void __launch_bounds__(256)
layer2_fused_kernel(const _Float16* __restrict__ s_in,
                    const _Float16* __restrict__ wpack,
                    const float* __restrict__ b2,
                    const float* __restrict__ lif_w,
                    const float* __restrict__ lif_b,
                    _Float16* __restrict__ h1_out) {
    extern __shared__ char smem[];
    _Float16*      s_tile = (_Float16*)smem;                          // 10*20*20 halfs
    float*         drives = (float*)(smem + 8192);                    // 336*32 f32
    float*         c2mem  = (float*)(smem + 51200);                   // 10*18*18*4 f32
    unsigned char* c2spk  = (unsigned char*)(smem + 103040);          // 12960 B
    float*         h1s    = (float*)(smem + 116096);                  // 3240 f32
    _Float16*      bfr    = (_Float16*)(smem + 129056);               // 3072 halfs
    float*         b2s    = (float*)(smem + 135200);                  // 30 f32 (4*b2)

    int tid  = threadIdx.x;
    int b    = blockIdx.x;
    int lane = tid & 31, wave = tid >> 5;
    int hi   = lane >> 4;

    for (int i = tid; i < 4000;  i += 256) s_tile[i] = (_Float16)0.f;  // halo zeros
    for (int i = tid; i < 12960; i += 256) { c2mem[i] = 0.f; c2spk[i] = 0; }
    for (int i = tid; i < 3240;  i += 256) h1s[i] = 0.f;
    for (int i = tid; i < 3072;  i += 256) bfr[i] = wpack[i];
    for (int i = tid; i < 30;    i += 256) b2s[i] = 4.f * b2[i];
    float lw0 = lif_w[0], lw1 = lif_w[1], lw2 = lif_w[2], lb = lif_b[0];

    // loop-invariant per-lane im2col offsets (compile-time except one cndmask)
    int aoff[3][16];
    #pragma unroll
    for (int kk = 0; kk < 3; ++kk)
        #pragma unroll
        for (int e = 0; e < 16; ++e) {
            const int ka = kk * 32 + (e < 8 ? e : e + 8);          // hi = 0
            const int kb = (ka + 8) < 90 ? (ka + 8) : 0;           // hi = 1 (clamped;
            aoff[kk][e] = hi ? OFFK(kb) : OFFK(ka);                //  B holds 0 there)
        }
    __syncthreads();

    for (int t = 0; t < TW; ++t) {
        // ---- stage s_t interior into padded LDS tile + prefetch s_{t+1} ----
        const _Float16* sg = s_in + (size_t)(b * TW + t) * 3240;
        for (int i = tid; i < 3240; i += 256) {
            int c = i / 324, q = i % 324;
            int y = q / 18, xx = q % 18;
            s_tile[c * 400 + (y + 1) * 20 + (xx + 1)] = sg[i];
            if (t + 1 < TW) __builtin_prefetch(sg + 3240 + i, 0, 3);
        }
        __syncthreads();

        // ---- implicit-GEMM conv: 21 M-tile jobs over 8 waves ---------------
        for (int mtile = wave; mtile < 21; mtile += 8) {
            int mstart = (mtile < 20) ? mtile * 16 : 308;   // overlapped last tile
            int mrow   = mstart + (lane & 15);
            int y = mrow / 18, xx = mrow % 18;
            const _Float16* srow = s_tile + (y * 20 + xx);
            v8f acc0 = {}, acc1 = {};
            #pragma unroll
            for (int kk = 0; kk < 3; ++kk) {
                v16h a;
                #pragma unroll
                for (int e = 0; e < 16; ++e) a[e] = srow[aoff[kk][e]];
                v16h b0 = *(const v16h*)(bfr + (kk * 32 + lane) * 16);
                v16h b1 = *(const v16h*)(bfr + ((3 + kk) * 32 + lane) * 16);
                acc0 = __builtin_amdgcn_wmma_f32_16x16x32_f16(
                           false, a, false, b0, (short)0, acc0, false, false);
                acc1 = __builtin_amdgcn_wmma_f32_16x16x32_f16(
                           false, a, false, b1, (short)0, acc1, false, false);
            }
            int nlo = lane & 15;
            #pragma unroll
            for (int r = 0; r < 8; ++r) {
                int m = mstart + r + 8 * hi;  // C/D layout (ISA 7.12.2)
                drives[m * 32 + nlo]      = acc0[r];
                drives[m * 32 + 16 + nlo] = acc1[r];
            }
        }
        __syncthreads();

        // ---- LIF update + 2x2 pool + h1 accumulation ----------------------
        for (int jb = tid; jb < 810; jb += 256) {
            int c = jb / 81, q = jb % 81;
            int by = q / 9, bx = q % 9;
            float pooled[4] = {0.f, 0.f, 0.f, 0.f};
            #pragma unroll
            for (int p = 0; p < 4; ++p) {
                int py = 2 * by + (p >> 1), px = 2 * bx + (p & 1);
                int base  = ((c * 18 + py) * 18 + px) * 4;
                float m0 = c2mem[base + 0], m1 = c2mem[base + 1], m2 = c2mem[base + 2];
                float inner = lw0 * m0 + lw1 * m1 + lw2 * m2 + lb;
                int dbase = (py * 18 + px) * 32;
                #pragma unroll
                for (int h = 0; h < 4; ++h) {
                    float d = (h < 3) ? (drives[dbase + h * 10 + c] + b2s[h * 10 + c])
                                      : inner;
                    float mo = c2mem[base + h];
                    float sp = (float)c2spk[base + h];
                    float mn = mo * DECAY * (1.f - sp) + d;
                    float sn = (mn > THRESH) ? 1.f : 0.f;
                    c2mem[base + h] = mn;
                    c2spk[base + h] = (unsigned char)(mn > THRESH);
                    pooled[h] += sn;
                }
            }
            int hb = jb * 4;          // flatten (10,9,9,4)
            #pragma unroll
            for (int h = 0; h < 4; ++h) h1s[hb + h] += pooled[h] * 0.25f;
        }
        __syncthreads();
    }

    // h1 entries are exact multiples of 0.25 in [0,10] -> lossless f16
    for (int i = tid; i < 3240; i += 256)
        h1_out[b * 3240 + i] = (_Float16)h1s[i];
}

// ---------------------------------------------------------------------------
// Kernel 3: FC GEMM (256 x 3240) @ (3240 x 50) via f16 WMMA, one wave per
// 16x16 output tile. 101 guard-free K-chunks + 1 masked tail chunk.
// The 1/time_window scale is folded into the (rounded anyway) f16 B weights;
// A (h1) is lossless f16. Out-of-range B columns are only clamped for address
// safety: they pollute only C columns >= 50, which are never stored.
// ---------------------------------------------------------------------------
__global__ void __launch_bounds__(32)
head_fc_kernel(const _Float16* __restrict__ h1,
               const float* __restrict__ fc_w,
               const float* __restrict__ fc_b,
               float* __restrict__ outs) {
    int lane  = threadIdx.x;
    int mtile = blockIdx.x >> 2;
    int ntile = blockIdx.x & 3;
    int hi    = lane >> 4;
    int mrow  = mtile * 16 + (lane & 15);
    int ncol  = ntile * 16 + (lane & 15);
    int ncolc = ncol < 50 ? ncol : 49;

    int kloc[16];
    #pragma unroll
    for (int e = 0; e < 16; ++e) kloc[e] = (e < 8 ? e : e + 8) + 8 * hi;

    const _Float16* arow = h1 + (size_t)mrow * 3240;
    const float*    brow = fc_w + (size_t)ncolc * 3240;

    v8f acc = {};
    for (int kc = 0; kc < 101; ++kc) {        // k <= 3231 : no bounds checks
        int kbase = kc * 32;
        v16h a, bb;
        #pragma unroll
        for (int e = 0; e < 16; ++e) {
            int k = kbase + kloc[e];
            a[e]  = arow[k];
            bb[e] = (_Float16)(brow[k] * (1.f / TW));
        }
        acc = __builtin_amdgcn_wmma_f32_16x16x32_f16(
                  false, a, false, bb, (short)0, acc, false, false);
    }
    {   // tail chunk kbase = 3232: valid only for hi==0 && e<8
        v16h a, bb;
        #pragma unroll
        for (int e = 0; e < 16; ++e) {
            if (e < 8) {
                int k = 3232 + e;                               // < 3240, safe
                _Float16 av = arow[k];
                _Float16 bv = (_Float16)(brow[k] * (1.f / TW));
                a[e]  = hi ? (_Float16)0.f : av;
                bb[e] = hi ? (_Float16)0.f : bv;
            } else {
                a[e] = (_Float16)0.f; bb[e] = (_Float16)0.f;
            }
        }
        acc = __builtin_amdgcn_wmma_f32_16x16x32_f16(
                  false, a, false, bb, (short)0, acc, false, false);
    }

    #pragma unroll
    for (int r = 0; r < 8; ++r) {
        int m = mtile * 16 + r + 8 * hi;
        if (ncol < 50) outs[m * 50 + ncol] = acc[r] + fc_b[ncol];
    }
}

// ---------------------------------------------------------------------------
// Kernel 4: task head  out[b,tt,o] = outs[b,:] . task_w[tt,o,:] + task_b[tt,o]
// ---------------------------------------------------------------------------
__global__ void task_head_kernel(const float* __restrict__ outs,
                                 const float* __restrict__ task_w,
                                 const float* __restrict__ task_b,
                                 float* __restrict__ out) {
    int idx = blockIdx.x * blockDim.x + threadIdx.x;
    if (idx >= 256 * 30) return;
    int b = idx / 30, r = idx % 30;
    float acc = task_b[r];
    const float* tw = task_w + r * 50;
    const float* ob = outs + b * 50;
    #pragma unroll 10
    for (int f = 0; f < 50; ++f) acc += ob[f] * tw[f];
    out[idx] = acc;
}

// ---------------------------------------------------------------------------
// Workspace layout (bytes):
//   [0, 16588800)            s      f16  (256,10,10,18,18)
//   [16588800, 16594944)     wpack  f16  3072
//   [16594944, 18253824)     h1     f16  (256,3240)
//   [18253824, 18305024)     outs   f32  (256,50)
// ---------------------------------------------------------------------------
extern "C" void kernel_launch(void* const* d_in, const int* in_sizes, int n_in,
                              void* d_out, int out_size, void* d_ws, size_t ws_size,
                              hipStream_t stream) {
    const float* x      = (const float*)d_in[0];
    const float* w1     = (const float*)d_in[1];
    const float* b1     = (const float*)d_in[2];
    const float* w2     = (const float*)d_in[3];
    const float* b2     = (const float*)d_in[4];
    const float* lif_w  = (const float*)d_in[5];
    const float* lif_b  = (const float*)d_in[6];
    const float* fc_w   = (const float*)d_in[7];
    const float* fc_b   = (const float*)d_in[8];
    const float* task_w = (const float*)d_in[9];
    const float* task_b = (const float*)d_in[10];
    float* out = (float*)d_out;
    char*  ws  = (char*)d_ws;

    _Float16* s_buf = (_Float16*)ws;
    _Float16* wpack = (_Float16*)(ws + 16588800);
    _Float16* h1    = (_Float16*)(ws + 16594944);
    float*    outs  = (float*)(ws + 18253824);

    pack_w2_kernel<<<12, 256, 0, stream>>>(w2, wpack);
    layer1_lif_kernel<<<3240, 256, 0, stream>>>(x, w1, b1, lif_w, lif_b, s_buf);
    layer2_fused_kernel<<<256, 256, 135328, stream>>>(s_buf, wpack, b2, lif_w, lif_b, h1);
    head_fc_kernel<<<64, 32, 0, stream>>>(h1, fc_w, fc_b, outs);
    task_head_kernel<<<30, 256, 0, stream>>>(outs, task_w, task_b, out);
}
